// FasterRCNNProposalGenerator_53042846105700
// MI455X (gfx1250) — compile-verified, matrix operand order checked by reading, not run on hardware
//
#include <hip/hip_runtime.h>
#include <math.h>

// ---------------- problem constants ----------------
#define BATCH   2
#define CIN     1024
#define HH      50
#define WW      76
#define MPIX    (HH*WW)            // 3800
#define NA      15
#define NANCH   (MPIX*NA)          // 57000
#define PRE_NMSN 2000
#define POST_NMSN 300
#define NROI    (BATCH*POST_NMSN)  // 600
#define NCLS    1600
#define NCLS1   1601
#define NPADC   1664               // 104 n-tiles * 16
#define NTILES_CLS 104
#define DET_PER 100
#define NDET    (BATCH*DET_PER)    // 200
#define CLIPLOG 4.135166556742356f
#define RPN_T   0.7f
#define DET_T   0.5f
#define SCORE_T 0.05f
#define NEG_INF (-3.402823466e38f)

#define MTILES_FC 38               // ceil(600/16)
#define WFRAG_FRAGS (9*32*64*32)   // 589824 fragments of 16 bf16
#define CLSFRAG_FRAGS (32*NTILES_CLS*32)   // 106496
#define POOLFRAG_FRAGS (MTILES_FC*32*32)   // 38912

typedef __attribute__((ext_vector_type(16))) __bf16 v16bf;
typedef __attribute__((ext_vector_type(8)))  __bf16 v8bf;
typedef __attribute__((ext_vector_type(8)))  float  v8f;

// K index within a 32-wide chunk for element e (0..15) of a lane's WMMA fragment.
// (ISA 7.12.2, 16-bit A 16x32 / mirrored B): lanes>=16 use K half +8; VGPR j<4 -> K base 0, j>=4 -> +16.
// Note: elements 0..7 are the contiguous run khalf..khalf+7; elements 8..15 are 16+khalf..16+khalf+7.
__device__ __forceinline__ int frag_k(int lane, int e) {
  int j = e >> 1, p = e & 1;
  return ((j < 4) ? 0 : 16) + ((lane >> 4) << 3) + ((j & 3) << 1) + p;
}

__device__ __forceinline__ float wave_reduce_sum(float v) {
#pragma unroll
  for (int o = 16; o > 0; o >>= 1) v += __shfl_down(v, o, 32);
  return v;
}

__device__ __forceinline__ float iou_box(float ax1, float ay1, float ax2, float ay2,
                                         float bx1, float by1, float bx2, float by2) {
  float aa = (ax2 - ax1) * (ay2 - ay1);
  float ab = (bx2 - bx1) * (by2 - by1);
  float ix = fmaxf(fminf(ax2, bx2) - fmaxf(ax1, bx1), 0.f);
  float iy = fmaxf(fminf(ay2, by2) - fmaxf(ay1, by1), 0.f);
  float inter = ix * iy;
  return inter / (aa + ab - inter + 1e-9f);
}

// ---------------- P0: feature map NCHW fp32 -> NHWC bf16 ----------------
__global__ void featbf_prep_kernel(const float* __restrict__ feat, __bf16* __restrict__ featbf) {
  int pos = blockIdx.x * 256 + threadIdx.x;
  if (pos >= BATCH * MPIX * CIN) return;
  int c = pos & (CIN - 1);
  int m = (pos >> 10) % MPIX;
  int b = (pos >> 10) / MPIX;
  featbf[pos] = (__bf16)feat[((size_t)b * CIN + c) * MPIX + m];
}

// ---------------- P1: conv weights -> bf16 WMMA B-fragment order ----------------
// layout: fid = ((kyx*32 + cc)*64 + nt)*32 + lane ; 16 contiguous bf16 per fragment
__global__ void wfrag_prep_kernel(const float* __restrict__ wconv, __bf16* __restrict__ wfrag) {
  int fid = blockIdx.x * 256 + threadIdx.x;
  if (fid >= WFRAG_FRAGS) return;
  int lane = fid & 31;
  int nt   = (fid >> 5) & 63;
  int cc   = (fid >> 11) & 31;
  int kyx  = fid >> 16;
  int n = nt * 16 + (lane & 15);
  v16bf fr;
#pragma unroll
  for (int e = 0; e < 16; ++e) {
    int c = cc * 32 + frag_k(lane, e);
    fr[e] = (__bf16)wconv[(((size_t)n * CIN + c) * 3 + (kyx / 3)) * 3 + (kyx % 3)];
  }
  *((v16bf*)(wfrag + (size_t)fid * 16)) = fr;
}

// ---------------- P2: cls_w -> bf16 WMMA B-fragment order (transposed, N padded) ----------------
// layout: fid = (cc*104 + nt)*32 + lane
__global__ void clsfrag_prep_kernel(const float* __restrict__ clsw, __bf16* __restrict__ clsfrag) {
  int fid = blockIdx.x * 256 + threadIdx.x;
  if (fid >= CLSFRAG_FRAGS) return;
  int lane = fid & 31;
  int tmp  = fid >> 5;
  int nt   = tmp % NTILES_CLS;
  int cc   = tmp / NTILES_CLS;
  int n = nt * 16 + (lane & 15);
  v16bf fr;
#pragma unroll
  for (int e = 0; e < 16; ++e) {
    int c = cc * 32 + frag_k(lane, e);
    fr[e] = (__bf16)((n < NCLS1) ? clsw[(size_t)n * CIN + c] : 0.f);
  }
  *((v16bf*)(clsfrag + (size_t)fid * 16)) = fr;
}

// ---------------- P3: pooled -> bf16 WMMA A-fragment order ----------------
// layout: fid = (mt*32 + cc)*32 + lane
__global__ void poolfrag_prep_kernel(const float* __restrict__ pooled, __bf16* __restrict__ poolfrag) {
  int fid = blockIdx.x * 256 + threadIdx.x;
  if (fid >= POOLFRAG_FRAGS) return;
  int lane = fid & 31;
  int cc   = (fid >> 5) & 31;
  int mt   = fid >> 10;
  int m = mt * 16 + (lane & 15);
  v16bf fr;
#pragma unroll
  for (int e = 0; e < 16; ++e) {
    int c = cc * 32 + frag_k(lane, e);
    fr[e] = (__bf16)((m < NROI) ? pooled[(size_t)m * CIN + c] : 0.f);
  }
  *((v16bf*)(poolfrag + (size_t)fid * 16)) = fr;
}

// ---------------- K1: 3x3 RPN conv, implicit GEMM, bf16 WMMA ----------------
// grid (ceil(3800/32)=119, 8, 2), block 256 (8 waves). Block tile: 32(M) x 128(N).
// Wave: one 16-wide N tile, two M subtiles -> 2 WMMA per K-chunk sharing one B fragment.
// B fragments stream from L2 (pre-swizzled); A fragments double-buffered in LDS.
// A staging: per fragment = two contiguous 16B loads from NHWC bf16 + two 16B LDS stores.
__global__ __launch_bounds__(256) void rpn_conv_kernel(const __bf16* __restrict__ featbf,
                                                       const __bf16* __restrict__ wfrag,
                                                       const float* __restrict__ bconv,
                                                       float* __restrict__ t) {
  const int b = blockIdx.z;
  const int m0 = blockIdx.x * 32;
  const int tid = threadIdx.x;
  const int wv = tid >> 5, lane = tid & 31;
  const int nt = blockIdx.y * 8 + wv;    // global n-tile 0..63

  __shared__ __bf16 lA[2][2 * 32 * 16];  // [parity][subtile*32 + lane][16]

  v8f acc0 = {}, acc1 = {};

  // per-thread staging invariants (64 staging threads)
  const int sub = tid >> 5, l2 = tid & 31;
  const int sm = m0 + sub * 16 + (l2 & 15);
  const int skhalf = (l2 >> 4) << 3;
  const int sy0 = sm / WW, sx0 = sm % WW;
  const bool mok = sm < MPIX;

  int kc = 0;
  for (int kyx = 0; kyx < 9; ++kyx) {
    const int ky = kyx / 3 - 1, kx = kyx % 3 - 1;
    for (int cc = 0; cc < 32; ++cc, ++kc) {
      const int par = kc & 1;
      if (tid < 64) {
        int y = sy0 + ky, x = sx0 + kx;
        bool ok = mok && (y >= 0) && (y < HH) && (x >= 0) && (x < WW);
        __bf16* dst = &lA[par][(sub * 32 + l2) * 16];
        if (ok) {
          const __bf16* p = featbf + (((size_t)b * MPIX + y * WW + x) * CIN) + cc * 32 + skhalf;
          v8bf lo = *((const v8bf*)p);
          v8bf hi = *((const v8bf*)(p + 16));
          *((v8bf*)dst) = lo;
          *((v8bf*)(dst + 8)) = hi;
        } else {
          v8bf z = {};
          *((v8bf*)dst) = z;
          *((v8bf*)(dst + 8)) = z;
        }
      }
      __syncthreads();
      if (kc + 1 < 288)
        __builtin_prefetch(wfrag + (((size_t)(kc + 1) * 64 + nt) * 32 + lane) * 16, 0, 1);
      const v16bf bfr = *((const v16bf*)(wfrag + (((size_t)kc * 64 + nt) * 32 + lane) * 16));
      v16bf af0 = *((const v16bf*)&lA[par][lane * 16]);
      v16bf af1 = *((const v16bf*)&lA[par][(32 + lane) * 16]);
      acc0 = __builtin_amdgcn_wmma_f32_16x16x32_bf16(false, af0, false, bfr,
                                                     (short)0, acc0, false, false);
      acc1 = __builtin_amdgcn_wmma_f32_16x16x32_bf16(false, af1, false, bfr,
                                                     (short)0, acc1, false, false);
    }
  }
  // epilogue: bias + ReLU; t stored [b][pixel][channel]
  int n = nt * 16 + (lane & 15);
  float bias = bconv[n];
#pragma unroll
  for (int v = 0; v < 8; ++v) {
    int mrow = v + ((lane >> 4) << 3);
    int m = m0 + mrow;
    if (m < MPIX) {
      float r = acc0[v] + bias;
      t[((size_t)b * MPIX + m) * CIN + n] = r > 0.f ? r : 0.f;
    }
    m = m0 + 16 + mrow;
    if (m < MPIX) {
      float r = acc1[v] + bias;
      t[((size_t)b * MPIX + m) * CIN + n] = r > 0.f ? r : 0.f;
    }
  }
}

// ---------------- K2: objectness + delta 1x1 heads (wave dot products) ----------------
__global__ __launch_bounds__(256) void objdelta_kernel(const float* __restrict__ t,
                                                       const float* __restrict__ objw,
                                                       const float* __restrict__ objb,
                                                       const float* __restrict__ dltw,
                                                       const float* __restrict__ dltb,
                                                       float* __restrict__ scores,
                                                       float* __restrict__ deltas) {
  int gm = blockIdx.x;
  int b = gm / MPIX, m = gm % MPIX;
  const float* tr = t + (size_t)gm * CIN;
  int wv = threadIdx.x >> 5, lane = threadIdx.x & 31;
  for (int o = wv; o < 75; o += 8) {
    const float* wr = (o < NA) ? (objw + (size_t)o * CIN) : (dltw + (size_t)(o - NA) * CIN);
    float s = 0.f;
    for (int c = lane; c < CIN; c += 32) s += tr[c] * wr[c];
    s = wave_reduce_sum(s);
    if (lane == 0) {
      if (o < NA) scores[(size_t)b * NANCH + m * NA + o] = s + objb[o];
      else {
        int d = o - NA;
        deltas[((size_t)b * NANCH + m * NA + d / 4) * 4 + (d & 3)] = s + dltb[d];
      }
    }
  }
}

// ---------------- K3: anchors + apply_deltas + clip ----------------
__global__ void proposal_kernel(const float* __restrict__ deltas, const int* __restrict__ sizes,
                                float* __restrict__ props) {
  int i = blockIdx.x * 256 + threadIdx.x;
  if (i >= BATCH * NANCH) return;
  int b = i / NANCH, ii = i % NANCH;
  int pix = ii / NA, a = ii % NA;
  int y = pix / WW, x = pix % WW;
  const float SZ[5] = {32.f, 64.f, 128.f, 256.f, 512.f};
  const float RT[3] = {0.5f, 1.f, 2.f};
  float s = SZ[a / 3], r = RT[a % 3];
  float wa = s * sqrtf(1.f / r);
  float ha = wa * r;
  float cx = x * 16.f, cy = y * 16.f;
  const float* d = deltas + (size_t)i * 4;
  float dw = fminf(d[2], CLIPLOG), dh = fminf(d[3], CLIPLOG);
  float pcx = d[0] * wa + cx, pcy = d[1] * ha + cy;
  float pw = expf(dw) * wa, ph = expf(dh) * ha;
  float Hi = (float)sizes[b * 2], Wi = (float)sizes[b * 2 + 1];
  float* o = props + (size_t)i * 4;
  o[0] = fminf(fmaxf(pcx - 0.5f * pw, 0.f), Wi);
  o[1] = fminf(fmaxf(pcy - 0.5f * ph, 0.f), Hi);
  o[2] = fminf(fmaxf(pcx + 0.5f * pw, 0.f), Wi);
  o[3] = fminf(fmaxf(pcy + 0.5f * ph, 0.f), Hi);
}

// ---------------- K4: top-k 2000 via iterative block argmax ----------------
__global__ void copy_scores_kernel(const float* __restrict__ s, float* __restrict__ d) {
  int i = blockIdx.x * 256 + threadIdx.x;
  if (i < BATCH * NANCH) d[i] = s[i];
}

__global__ __launch_bounds__(1024) void topk_kernel(float* __restrict__ scws,
                                                    int* __restrict__ tkidx,
                                                    float* __restrict__ tksc) {
  int b = blockIdx.x;
  float* sc = scws + (size_t)b * NANCH;
  __shared__ float smax[1024];
  __shared__ int sidx[1024];
  for (int tsel = 0; tsel < PRE_NMSN; ++tsel) {
    float best = NEG_INF;
    int bi = 0;
    for (int i = threadIdx.x; i < NANCH; i += 1024) {
      float v = sc[i];
      if (v > best) { best = v; bi = i; }
    }
    smax[threadIdx.x] = best; sidx[threadIdx.x] = bi;
    __syncthreads();
    for (int s = 512; s > 0; s >>= 1) {
      if (threadIdx.x < s && smax[threadIdx.x + s] > smax[threadIdx.x]) {
        smax[threadIdx.x] = smax[threadIdx.x + s];
        sidx[threadIdx.x] = sidx[threadIdx.x + s];
      }
      __syncthreads();
    }
    if (threadIdx.x == 0) {
      tkidx[b * PRE_NMSN + tsel] = sidx[0];
      tksc[b * PRE_NMSN + tsel] = smax[0];
      sc[sidx[0]] = NEG_INF;
    }
    __syncthreads();
  }
}

__global__ void gather_topk_kernel(const float* __restrict__ props, const int* __restrict__ tkidx,
                                   float* __restrict__ tb) {
  int i = blockIdx.x * 256 + threadIdx.x;
  if (i >= BATCH * PRE_NMSN) return;
  int b = i / PRE_NMSN;
  const float* p = props + ((size_t)b * NANCH + tkidx[i]) * 4;
  float* o = tb + (size_t)i * 4;
  o[0] = p[0]; o[1] = p[1]; o[2] = p[2]; o[3] = p[3];
}

// ---------------- K5: RPN NMS ----------------
__global__ __launch_bounds__(256) void rpn_nms_kernel(const float* __restrict__ tb,
                                                      float* __restrict__ pboxes) {
  int b = blockIdx.x;
  __shared__ int keep[PRE_NMSN];
  for (int i = threadIdx.x; i < PRE_NMSN; i += 256) keep[i] = 1;
  __syncthreads();
  const float* bx = tb + (size_t)b * PRE_NMSN * 4;
  for (int i = 0; i < PRE_NMSN; ++i) {
    __syncthreads();
    if (!keep[i]) continue;           // uniform (LDS read after barrier)
    float ax1 = bx[i * 4 + 0], ay1 = bx[i * 4 + 1], ax2 = bx[i * 4 + 2], ay2 = bx[i * 4 + 3];
    for (int j = i + 1 + threadIdx.x; j < PRE_NMSN; j += 256) {
      if (!keep[j]) continue;
      if (iou_box(ax1, ay1, ax2, ay2, bx[j * 4], bx[j * 4 + 1], bx[j * 4 + 2], bx[j * 4 + 3]) > RPN_T)
        keep[j] = 0;
    }
  }
  __syncthreads();
  if (threadIdx.x == 0) {
    int cnt = 0;
    for (int i = 0; i < PRE_NMSN && cnt < POST_NMSN; ++i)
      if (keep[i]) {
        float* o = pboxes + ((size_t)b * POST_NMSN + cnt) * 4;
        o[0] = bx[i * 4]; o[1] = bx[i * 4 + 1]; o[2] = bx[i * 4 + 2]; o[3] = bx[i * 4 + 3];
        ++cnt;
      }
    for (int i = 0; i < PRE_NMSN && cnt < POST_NMSN; ++i)
      if (!keep[i]) {
        float* o = pboxes + ((size_t)b * POST_NMSN + cnt) * 4;
        o[0] = bx[i * 4]; o[1] = bx[i * 4 + 1]; o[2] = bx[i * 4 + 2]; o[3] = bx[i * 4 + 3];
        ++cnt;
      }
  }
}

// ---------------- K6: ROI align (7x7, mean pooled) ----------------
__global__ __launch_bounds__(256) void roi_kernel(const float* __restrict__ feat,
                                                  const float* __restrict__ pboxes,
                                                  float* __restrict__ pooled) {
  int roi = blockIdx.x;
  int b = roi / POST_NMSN;
  const float* bx = pboxes + (size_t)roi * 4;
  __shared__ int xi0[7], xi1[7], yi0[7], yi1[7];
  __shared__ float lxa[7], lya[7];
  const float S = 1.f / 16.f;
  if (threadIdx.x < 7) {
    int p = threadIdx.x;
    float x1 = bx[0] * S, x2 = bx[2] * S;
    float g = (p + 0.5f) / 7.f;
    float xs = x1 + g * (x2 - x1) - 0.5f;
    float x0f = floorf(xs);
    lxa[p] = xs - x0f;
    int x0i = (int)x0f; x0i = x0i < 0 ? 0 : (x0i > WW - 1 ? WW - 1 : x0i);
    xi0[p] = x0i; xi1[p] = (x0i + 1 > WW - 1) ? WW - 1 : x0i + 1;
  } else if (threadIdx.x < 14) {
    int p = threadIdx.x - 7;
    float y1 = bx[1] * S, y2 = bx[3] * S;
    float g = (p + 0.5f) / 7.f;
    float ys = y1 + g * (y2 - y1) - 0.5f;
    float y0f = floorf(ys);
    lya[p] = ys - y0f;
    int y0i = (int)y0f; y0i = y0i < 0 ? 0 : (y0i > HH - 1 ? HH - 1 : y0i);
    yi0[p] = y0i; yi1[p] = (y0i + 1 > HH - 1) ? HH - 1 : y0i + 1;
  }
  __syncthreads();
  for (int c = threadIdx.x; c < CIN; c += 256) {
    const float* f = feat + ((size_t)b * CIN + c) * HH * WW;
    float sum = 0.f;
#pragma unroll
    for (int py = 0; py < 7; ++py)
#pragma unroll
      for (int px = 0; px < 7; ++px) {
        float ly = lya[py], lx = lxa[px];
        float v00 = f[yi0[py] * WW + xi0[px]];
        float v01 = f[yi0[py] * WW + xi1[px]];
        float v10 = f[yi1[py] * WW + xi0[px]];
        float v11 = f[yi1[py] * WW + xi1[px]];
        sum += v00 * (1.f - ly) * (1.f - lx) + v01 * (1.f - ly) * lx +
               v10 * ly * (1.f - lx) + v11 * ly * lx;
      }
    pooled[(size_t)roi * CIN + c] = sum / 49.f;
  }
}

// ---------------- K7: classifier FC (600x1024 @ 1024x1601), pure-streaming WMMA ----------------
// grid (38, 13), block 256 (8 waves). No LDS, no barriers: both operands pre-swizzled.
__global__ __launch_bounds__(256) void fc_cls_kernel(const __bf16* __restrict__ poolfrag,
                                                     const __bf16* __restrict__ clsfrag,
                                                     const float* __restrict__ clsb,
                                                     float* __restrict__ logits) {
  const int mt = blockIdx.x;
  const int wv = threadIdx.x >> 5, lane = threadIdx.x & 31;
  const int nt = blockIdx.y * 8 + wv;    // 0..103
  v8f acc = {};
  for (int cc = 0; cc < 32; ++cc) {
    if (cc + 1 < 32) {
      __builtin_prefetch(poolfrag + (((size_t)mt * 32 + cc + 1) * 32 + lane) * 16, 0, 1);
      __builtin_prefetch(clsfrag + (((size_t)(cc + 1) * NTILES_CLS + nt) * 32 + lane) * 16, 0, 1);
    }
    v16bf af = *((const v16bf*)(poolfrag + (((size_t)mt * 32 + cc) * 32 + lane) * 16));
    v16bf bfr = *((const v16bf*)(clsfrag + (((size_t)cc * NTILES_CLS + nt) * 32 + lane) * 16));
    acc = __builtin_amdgcn_wmma_f32_16x16x32_bf16(false, af, false, bfr,
                                                  (short)0, acc, false, false);
  }
  int n = nt * 16 + (lane & 15);
  if (n < NCLS1) {
    float bias = clsb[n];
#pragma unroll
    for (int v = 0; v < 8; ++v) {
      int m = mt * 16 + v + ((lane >> 4) << 3);
      if (m < NROI) logits[(size_t)m * NPADC + n] = acc[v] + bias;
    }
  }
}

// ---------------- K8: softmax + per-row max foreground prob ----------------
__global__ __launch_bounds__(256) void softmax_kernel(const float* __restrict__ logits,
                                                      float* __restrict__ probs,
                                                      float* __restrict__ maxsc) {
  int row = blockIdx.x;
  const float* lg = logits + (size_t)row * NPADC;
  __shared__ float red[256];
  int tid = threadIdx.x;
  float mx = NEG_INF;
  for (int n = tid; n < NCLS1; n += 256) mx = fmaxf(mx, lg[n]);
  red[tid] = mx; __syncthreads();
  for (int s = 128; s > 0; s >>= 1) { if (tid < s) red[tid] = fmaxf(red[tid], red[tid + s]); __syncthreads(); }
  float M = red[0]; __syncthreads();
  float sm = 0.f;
  for (int n = tid; n < NCLS1; n += 256) sm += expf(lg[n] - M);
  red[tid] = sm; __syncthreads();
  for (int s = 128; s > 0; s >>= 1) { if (tid < s) red[tid] += red[tid + s]; __syncthreads(); }
  float S = red[0]; __syncthreads();
  float pm = NEG_INF;
  for (int n = tid; n < NCLS; n += 256) {
    float p = expf(lg[n] - M) / S;
    probs[(size_t)row * NCLS + n] = p;
    pm = fmaxf(pm, p);
  }
  red[tid] = pm; __syncthreads();
  for (int s = 128; s > 0; s >>= 1) { if (tid < s) red[tid] = fmaxf(red[tid], red[tid + s]); __syncthreads(); }
  if (tid == 0) maxsc[row] = red[0];
}

// ---------------- K9: detection box head + apply_deltas(10,10,5,5) + clip ----------------
__global__ __launch_bounds__(256) void detbox_kernel(const float* __restrict__ pooled,
                                                     const float* __restrict__ boxw,
                                                     const float* __restrict__ boxb,
                                                     const float* __restrict__ pboxes,
                                                     const int* __restrict__ sizes,
                                                     float* __restrict__ detb) {
  int m = blockIdx.x;
  int wv = threadIdx.x >> 5, lane = threadIdx.x & 31;
  __shared__ float d[4];
  if (wv < 4) {
    float s = 0.f;
    for (int c = lane; c < CIN; c += 32) s += pooled[(size_t)m * CIN + c] * boxw[(size_t)wv * CIN + c];
    s = wave_reduce_sum(s);
    if (lane == 0) d[wv] = s + boxb[wv];
  }
  __syncthreads();
  if (threadIdx.x == 0) {
    const float* pb = pboxes + (size_t)m * 4;
    float w = pb[2] - pb[0], h = pb[3] - pb[1];
    float cx = pb[0] + 0.5f * w, cy = pb[1] + 0.5f * h;
    float dx = d[0] / 10.f, dy = d[1] / 10.f;
    float dw = fminf(d[2] / 5.f, CLIPLOG), dh = fminf(d[3] / 5.f, CLIPLOG);
    float pcx = dx * w + cx, pcy = dy * h + cy;
    float pw = expf(dw) * w, ph = expf(dh) * h;
    int b = m / POST_NMSN;
    float Hi = (float)sizes[b * 2], Wi = (float)sizes[b * 2 + 1];
    float* o = detb + (size_t)m * 4;
    o[0] = fminf(fmaxf(pcx - 0.5f * pw, 0.f), Wi);
    o[1] = fminf(fmaxf(pcy - 0.5f * ph, 0.f), Hi);
    o[2] = fminf(fmaxf(pcx + 0.5f * pw, 0.f), Wi);
    o[3] = fminf(fmaxf(pcy + 0.5f * ph, 0.f), Hi);
  }
}

// ---------------- K10: detection NMS (300 boxes, serial per image) ----------------
__global__ void det_nms_kernel(const float* __restrict__ detb, const float* __restrict__ maxsc,
                               int* __restrict__ r) {
  int b = blockIdx.x;
  if (threadIdx.x != 0) return;
  float sc2[POST_NMSN];
  int ord[POST_NMSN], keep[POST_NMSN];
  for (int i = 0; i < POST_NMSN; ++i) {
    float v = maxsc[b * POST_NMSN + i];
    sc2[i] = (v > SCORE_T) ? v : NEG_INF;
    keep[i] = 1;
  }
  for (int t2 = 0; t2 < POST_NMSN; ++t2) {       // selection sort (descending)
    int bi = 0; float bv = NEG_INF;
    for (int i = 0; i < POST_NMSN; ++i)
      if (sc2[i] > bv) { bv = sc2[i]; bi = i; }
    ord[t2] = bi; sc2[bi] = NEG_INF - 1.f;
  }
  for (int i = 0; i < POST_NMSN; ++i) {
    if (!keep[i]) continue;
    const float* a = detb + (size_t)(b * POST_NMSN + ord[i]) * 4;
    for (int j = i + 1; j < POST_NMSN; ++j) {
      if (!keep[j]) continue;
      const float* q = detb + (size_t)(b * POST_NMSN + ord[j]) * 4;
      if (iou_box(a[0], a[1], a[2], a[3], q[0], q[1], q[2], q[3]) > DET_T) keep[j] = 0;
    }
  }
  int cnt = 0;
  for (int i = 0; i < POST_NMSN && cnt < DET_PER; ++i)
    if (keep[i]) r[b * DET_PER + cnt++] = b * POST_NMSN + ord[i];
  for (int i = 0; i < POST_NMSN && cnt < DET_PER; ++i)
    if (!keep[i]) r[b * DET_PER + cnt++] = b * POST_NMSN + ord[i];
}

// ---------------- K11: gather concatenated tuple output ----------------
__global__ void gather_out_kernel(const float* __restrict__ pooled, const float* __restrict__ pboxes,
                                  const float* __restrict__ probs, const int* __restrict__ r,
                                  float* __restrict__ out) {
  const int T0 = NDET * CIN;
  const int T1 = T0 + NDET * 4;
  const int T2 = T1 + NDET * NCLS;
  int pos = blockIdx.x * 256 + threadIdx.x;
  if (pos >= T2) return;
  if (pos < T0) {
    int i = pos >> 10, c = pos & 1023;
    out[pos] = pooled[(size_t)r[i] * CIN + c];
  } else if (pos < T1) {
    int q = pos - T0; int i = q >> 2, c = q & 3;
    out[pos] = pboxes[(size_t)r[i] * 4 + c];
  } else {
    int q = pos - T1; int i = q / NCLS, c = q % NCLS;
    out[pos] = probs[(size_t)r[i] * NCLS + c];
  }
}

// ---------------- host launcher ----------------
extern "C" void kernel_launch(void* const* d_in, const int* in_sizes, int n_in,
                              void* d_out, int out_size, void* d_ws, size_t ws_size,
                              hipStream_t stream) {
  (void)in_sizes; (void)n_in; (void)out_size; (void)ws_size;
  const int*   image_sizes = (const int*)d_in[1];
  const float* feat  = (const float*)d_in[2];
  const float* rpnw  = (const float*)d_in[3];
  const float* rpnb  = (const float*)d_in[4];
  const float* objw  = (const float*)d_in[5];
  const float* objb  = (const float*)d_in[6];
  const float* dltw  = (const float*)d_in[7];
  const float* dltb  = (const float*)d_in[8];
  const float* clsw  = (const float*)d_in[9];
  const float* clsb  = (const float*)d_in[10];
  const float* boxw  = (const float*)d_in[11];
  const float* boxb  = (const float*)d_in[12];
  float* out = (float*)d_out;

  char* ws = (char*)d_ws;
  size_t off = 0;
  auto alloc = [&](size_t bytes) -> char* {
    size_t o = (off + 255) & ~(size_t)255;
    off = o + bytes;
    return ws + o;
  };
  float*  t        = (float*) alloc((size_t)BATCH * MPIX * CIN * 4);
  __bf16* featbf   = (__bf16*)alloc((size_t)BATCH * MPIX * CIN * 2);
  __bf16* wfrag    = (__bf16*)alloc((size_t)WFRAG_FRAGS * 16 * 2);
  __bf16* clsfrag  = (__bf16*)alloc((size_t)CLSFRAG_FRAGS * 16 * 2);
  __bf16* poolfrag = (__bf16*)alloc((size_t)POOLFRAG_FRAGS * 16 * 2);
  float*  scores   = (float*) alloc((size_t)BATCH * NANCH * 4);
  float*  deltas   = (float*) alloc((size_t)BATCH * NANCH * 4 * 4);
  float*  props    = (float*) alloc((size_t)BATCH * NANCH * 4 * 4);
  float*  scws     = (float*) alloc((size_t)BATCH * NANCH * 4);
  int*    tkidx    = (int*)   alloc((size_t)BATCH * PRE_NMSN * 4);
  float*  tksc     = (float*) alloc((size_t)BATCH * PRE_NMSN * 4);
  float*  tb       = (float*) alloc((size_t)BATCH * PRE_NMSN * 4 * 4);
  float*  pboxes   = (float*) alloc((size_t)NROI * 4 * 4);
  float*  pooled   = (float*) alloc((size_t)NROI * CIN * 4);
  float*  logits   = (float*) alloc((size_t)NROI * NPADC * 4);
  float*  probs    = (float*) alloc((size_t)NROI * NCLS * 4);
  float*  maxsc    = (float*) alloc((size_t)NROI * 4);
  float*  detb     = (float*) alloc((size_t)NROI * 4 * 4);
  int*    ridx     = (int*)   alloc((size_t)NDET * 4);

  // P0: NCHW fp32 -> NHWC bf16 feature map
  featbf_prep_kernel<<<(BATCH * MPIX * CIN + 255) / 256, 256, 0, stream>>>(feat, featbf);
  // P1/P2: weight pre-swizzle into WMMA fragment order
  wfrag_prep_kernel<<<(WFRAG_FRAGS + 255) / 256, 256, 0, stream>>>(rpnw, wfrag);
  clsfrag_prep_kernel<<<(CLSFRAG_FRAGS + 255) / 256, 256, 0, stream>>>(clsw, clsfrag);
  // 1. 3x3 conv + ReLU (WMMA implicit GEMM)
  rpn_conv_kernel<<<dim3((MPIX + 31) / 32, 8, BATCH), 256, 0, stream>>>(featbf, wfrag, rpnb, t);
  // 2. objectness + deltas
  objdelta_kernel<<<BATCH * MPIX, 256, 0, stream>>>(t, objw, objb, dltw, dltb, scores, deltas);
  // 3. proposals
  proposal_kernel<<<(BATCH * NANCH + 255) / 256, 256, 0, stream>>>(deltas, image_sizes, props);
  // 4. top-k 2000
  copy_scores_kernel<<<(BATCH * NANCH + 255) / 256, 256, 0, stream>>>(scores, scws);
  topk_kernel<<<BATCH, 1024, 0, stream>>>(scws, tkidx, tksc);
  gather_topk_kernel<<<(BATCH * PRE_NMSN + 255) / 256, 256, 0, stream>>>(props, tkidx, tb);
  // 5. RPN NMS -> 300 proposal boxes/image
  rpn_nms_kernel<<<BATCH, 256, 0, stream>>>(tb, pboxes);
  // 6. ROI align 7x7 mean
  roi_kernel<<<NROI, 256, 0, stream>>>(feat, pboxes, pooled);
  // P3: pooled activations -> A-fragment order
  poolfrag_prep_kernel<<<(POOLFRAG_FRAGS + 255) / 256, 256, 0, stream>>>(pooled, poolfrag);
  // 7. classifier GEMM (pure-streaming WMMA)
  fc_cls_kernel<<<dim3(MTILES_FC, 13), 256, 0, stream>>>(poolfrag, clsfrag, clsb, logits);
  // 8. softmax + max foreground prob
  softmax_kernel<<<NROI, 256, 0, stream>>>(logits, probs, maxsc);
  // 9. detection boxes
  detbox_kernel<<<NROI, 256, 0, stream>>>(pooled, boxw, boxb, pboxes, image_sizes, detb);
  // 10. detection NMS -> 100/image
  det_nms_kernel<<<BATCH, 32, 0, stream>>>(detb, maxsc, ridx);
  // 11. gather tuple output
  gather_out_kernel<<<(NDET * (CIN + 4 + NCLS) + 255) / 256, 256, 0, stream>>>(pooled, pboxes, probs, ridx, out);
}
